// VectorQuantizer_17377437680341
// MI455X (gfx1250) — matrix operand order repaired and checked by reading, not compiled, not run on hardware
//
#include <hip/hip_runtime.h>
#include <hip/hip_bf16.h>

typedef __attribute__((ext_vector_type(16))) __bf16 v16bf;
typedef __attribute__((ext_vector_type(8)))  __bf16 v8bf;
typedef __attribute__((ext_vector_type(8)))  float  v8f;

#define BB 32
#define CC 256
#define HH 64
#define WW 64
#define HWP (HH * WW)          // 4096
#define NN  (BB * HWP)         // 131072 rows
#define KK  1024               // codes
#define DD  CC                 // 256

#define MT 64                  // rows per block
#define CT 32                  // codes per tile iteration
#define NT (KK / CT)           // 32 code tiles
#define LDP 8                  // bf16 pad elements
#define XST (DD + LDP)         // 264 (row stride, 528 B -> 16B aligned, conflict-free)

__device__ __forceinline__ __bf16 f2bf(float f) {
    union { float f; unsigned u; } v; v.f = f;
    unsigned r = v.u + 0x7FFFu + ((v.u >> 16) & 1u);   // round-to-nearest-even
    unsigned short h = (unsigned short)(r >> 16);
    return __builtin_bit_cast(__bf16, h);
}

__device__ __forceinline__ unsigned short bfbits(float f) {
    union { float f; unsigned u; } v; v.f = f;
    unsigned r = v.u + 0x7FFFu + ((v.u >> 16) & 1u);
    return (unsigned short)(r >> 16);
}

// ---------------- kernel 0a: per-code squared norms ----------------
__global__ __launch_bounds__(256) void vq_enorm_kernel(const float* __restrict__ emb,
                                                       float* __restrict__ enorm) {
    int k = blockIdx.x * blockDim.x + threadIdx.x;
    if (k >= KK) return;
    const float4* row = (const float4*)(emb + (size_t)k * DD);
    float s = 0.f;
#pragma unroll 4
    for (int i = 0; i < DD / 4; ++i) {
        float4 v = row[i];
        s += v.x * v.x + v.y * v.y + v.z * v.z + v.w * v.w;
    }
    enorm[k] = s;
}

// ---------------- kernel 0b: codebook f32 -> bf16 (done once, L2-resident after) ------
__global__ __launch_bounds__(256) void vq_embh_kernel(const float* __restrict__ emb,
                                                      __bf16* __restrict__ embh) {
    int i = (blockIdx.x * blockDim.x + threadIdx.x) * 4;
    float4 v = *(const float4*)(emb + i);
    unsigned lo = ((unsigned)bfbits(v.y) << 16) | bfbits(v.x);
    unsigned hi = ((unsigned)bfbits(v.w) << 16) | bfbits(v.z);
    uint2 pk; pk.x = lo; pk.y = hi;
    *(uint2*)(embh + i) = pk;
}

// ---------------- kernel 1: WMMA distance GEMM + argmin ----------------
__global__ __launch_bounds__(256)
void vq_argmin_kernel(const float* __restrict__ x, const __bf16* __restrict__ embh,
                      const float* __restrict__ enorm, int* __restrict__ out_idx) {
    // LDS only for the X transpose (BCHW -> row-major bf16) and the code norms.
    __shared__ __align__(16) __bf16 Xs[MT * XST];    // 33792 B
    __shared__ float  EnS[KK];                       //  4096 B
    __shared__ float  rowMin[MT];
    __shared__ int    rowIdx[MT];

    const int tid   = threadIdx.x;
    const int lane  = tid & 31;
    const int wave  = tid >> 5;          // 0..7
    const int rg    = wave >> 1;         // row group 0..3 (16 rows each)
    const int cs    = wave & 1;          // code half within a CT tile
    const int n0    = blockIdx.x * MT;   // MT | HWP, so one batch image per tile
    const int b     = n0 / HWP;
    const int hw0   = n0 % HWP;

    for (int i = tid; i < KK; i += 256) EnS[i] = enorm[i];

    // Stage X tile: rows are consecutive hw at fixed (b,c) -> coalesced float4 loads
    {
        const int r4 = (tid & 15) * 4;       // hw offset within tile
        const int c0 = tid >> 4;             // 0..15
        const float* xb = x + (size_t)b * CC * HWP + hw0;
        for (int c = c0; c < CC; c += 16) {
            float4 v = *(const float4*)(xb + (size_t)c * HWP + r4);
            Xs[(r4 + 0) * XST + c] = f2bf(v.x);
            Xs[(r4 + 1) * XST + c] = f2bf(v.y);
            Xs[(r4 + 2) * XST + c] = f2bf(v.z);
            Xs[(r4 + 3) * XST + c] = f2bf(v.w);
        }
    }
    __syncthreads();

    const int col   = lane & 15;         // A-row M / B-col N
    const int khalf = lane >> 4;         // K-half select per ISA 16-bit layouts

    // A fragments are loop-invariant (this wave's 16 X rows): load once, keep in
    // registers (8 x v16bf = 64 VGPRs).
    v16bf afrag[8];
    {
        const __bf16* xrow = &Xs[(rg * 16 + col) * XST];
#pragma unroll
        for (int ks = 0; ks < 8; ++ks) {
            const int k0 = ks * 32;
            // A 16x32 bf16: lane m, elems 0..7 -> K=k0+8*khalf+e ; 8..15 -> K=k0+16+8*khalf+(e-8)
            v8bf a0 = *(const v8bf*)(xrow + k0 + 8 * khalf);
            v8bf a1 = *(const v8bf*)(xrow + k0 + 16 + 8 * khalf);
            afrag[ks] = __builtin_shufflevector(a0, a1, 0,1,2,3,4,5,6,7,8,9,10,11,12,13,14,15);
        }
    }

    float minv[8];
    int   mini[8];
#pragma unroll
    for (int v = 0; v < 8; ++v) { minv[v] = 3.0e38f; mini[v] = 0; }

    // Per-lane B fragment source: two 16B slices per k-step of one bf16 codebook row.
    // B 32x16 bf16: lane n holds col n, elems 0..7 -> K = k0 + 8*khalf + e ; 8..15 -> K = k0 + 16 + 8*khalf + (e-8).
    // embh is L2/WGP$-resident; loop is barrier-free so loads pipeline freely.
    const __bf16* brow = embh + (size_t)(cs * 16 + col) * DD + 8 * khalf;

#pragma unroll 2
    for (int kt = 0; kt < NT; ++kt) {
        const __bf16* br = brow + (size_t)kt * CT * DD;
        if (kt + 1 < NT) {  // pull next tile's rows toward the caches
            __builtin_prefetch(br + (size_t)CT * DD, 0, 1);
        }

        v8f acc = {0.f, 0.f, 0.f, 0.f, 0.f, 0.f, 0.f, 0.f};
#pragma unroll
        for (int ks = 0; ks < 8; ++ks) {
            const int k0 = ks * 32;
            v8bf b0 = *(const v8bf*)(br + k0);
            v8bf b1 = *(const v8bf*)(br + k0 + 16);
            v16bf bm = __builtin_shufflevector(b0, b1, 0,1,2,3,4,5,6,7,8,9,10,11,12,13,14,15);
            acc = __builtin_amdgcn_wmma_f32_16x16x32_bf16(
                false, afrag[ks], false, bm, (short)0, acc, false, false);
        }

        // dist(row, code) = ||e||^2 - 2*dot ; C layout: c[v] = row (v + 8*khalf), col (lane&15)
        const int codeg = kt * CT + cs * 16 + col;
        const float en = EnS[codeg];
#pragma unroll
        for (int v = 0; v < 8; ++v) {
            float d = en - 2.0f * acc[v];
            if (d < minv[v]) { minv[v] = d; mini[v] = codeg; }
        }
    }

    // Reduce across the 16 lanes that share the same rows (xor masks stay in 16-lane halves)
#pragma unroll
    for (int v = 0; v < 8; ++v) {
        float mv = minv[v]; int mi = mini[v];
#pragma unroll
        for (int off = 8; off >= 1; off >>= 1) {
            float ov = __shfl_xor(mv, off, 32);
            int   oi = __shfl_xor(mi, off, 32);
            if (ov < mv || (ov == mv && oi < mi)) { mv = ov; mi = oi; }
        }
        minv[v] = mv; mini[v] = mi;
    }

    // Merge the two code-halves (cs=0 / cs=1 waves cover the same rows)
    if (cs == 0 && col == 0) {
#pragma unroll
        for (int v = 0; v < 8; ++v) {
            const int r = rg * 16 + khalf * 8 + v;
            rowMin[r] = minv[v];
            rowIdx[r] = mini[v];
        }
    }
    __syncthreads();
    if (cs == 1 && col == 0) {
#pragma unroll
        for (int v = 0; v < 8; ++v) {
            const int r = rg * 16 + khalf * 8 + v;
            float ov = rowMin[r]; int oi = rowIdx[r];
            float mv = minv[v];   int mi = mini[v];
            if (ov < mv || (ov == mv && oi < mi)) { mv = ov; mi = oi; }
            out_idx[n0 + r] = mi;
        }
    }
}

// ---------------- kernel 2: gather codebook rows, write BCHW, partial loss ----------------
__global__ __launch_bounds__(256)
void vq_gather_kernel(const float* __restrict__ x, const float* __restrict__ emb,
                      const int* __restrict__ idx, float* __restrict__ out,
                      float* __restrict__ partial) {
    const int n  = blockIdx.x * blockDim.x + threadIdx.x;   // one flat row per thread
    const int b  = n / HWP;
    const int hw = n % HWP;
    const int code = idx[n];
    const float* er = emb + (size_t)code * DD;
    const float* xp = x   + (size_t)b * CC * HWP + hw;
    float*       op = out + (size_t)b * CC * HWP + hw;
    float s = 0.f;
#pragma unroll 4
    for (int c = 0; c < CC; c += 4) {
        float4 e4 = *(const float4*)(er + c);
        float q, xv, d;
        q = e4.x; xv = xp[(size_t)(c + 0) * HWP]; d = q - xv; s += d * d; op[(size_t)(c + 0) * HWP] = q;
        q = e4.y; xv = xp[(size_t)(c + 1) * HWP]; d = q - xv; s += d * d; op[(size_t)(c + 1) * HWP] = q;
        q = e4.z; xv = xp[(size_t)(c + 2) * HWP]; d = q - xv; s += d * d; op[(size_t)(c + 2) * HWP] = q;
        q = e4.w; xv = xp[(size_t)(c + 3) * HWP]; d = q - xv; s += d * d; op[(size_t)(c + 3) * HWP] = q;
    }
    __shared__ float red[256];
    red[threadIdx.x] = s;
    __syncthreads();
    for (int o = 128; o >= 1; o >>= 1) {
        if (threadIdx.x < o) red[threadIdx.x] += red[threadIdx.x + o];
        __syncthreads();
    }
    if (threadIdx.x == 0) partial[blockIdx.x] = red[0];
}

// ---------------- kernel 3: final loss reduction ----------------
__global__ __launch_bounds__(256)
void vq_loss_kernel(const float* __restrict__ partial, int nparts, float* __restrict__ out_loss) {
    __shared__ float red[256];
    float s = 0.f;
    for (int i = threadIdx.x; i < nparts; i += 256) s += partial[i];
    red[threadIdx.x] = s;
    __syncthreads();
    for (int o = 128; o >= 1; o >>= 1) {
        if (threadIdx.x < o) red[threadIdx.x] += red[threadIdx.x + o];
        __syncthreads();
    }
    // loss = q_latent + 0.25 * e_latent = 1.25 * mean((q - x)^2)
    if (threadIdx.x == 0) *out_loss = 1.25f * red[0] / (float)((size_t)NN * CC);
}

extern "C" void kernel_launch(void* const* d_in, const int* in_sizes, int n_in,
                              void* d_out, int out_size, void* d_ws, size_t ws_size,
                              hipStream_t stream) {
    (void)in_sizes; (void)n_in; (void)out_size; (void)ws_size;
    const float* x   = (const float*)d_in[0];   // [32,256,64,64]
    const float* emb = (const float*)d_in[1];   // [1024,256]
    float* out = (float*)d_out;                 // [32,256,64,64] ++ [1] loss

    // workspace layout
    float*  enorm   = (float*)d_ws;                                         // 1024 f32
    __bf16* embh    = (__bf16*)((char*)d_ws + 4096);                        // 1024x256 bf16
    int*    idx     = (int*)((char*)d_ws + 4096 + (size_t)KK * DD * 2);     // NN i32
    float*  partial = (float*)((char*)d_ws + 4096 + (size_t)KK * DD * 2
                                              + (size_t)NN * 4);            // NN/256 f32

    vq_enorm_kernel<<<KK / 256, 256, 0, stream>>>(emb, enorm);
    vq_embh_kernel<<<(KK * DD / 4) / 256, 256, 0, stream>>>(emb, embh);
    vq_argmin_kernel<<<NN / MT, 256, 0, stream>>>(x, embh, enorm, idx);
    vq_gather_kernel<<<NN / 256, 256, 0, stream>>>(x, emb, idx, out, partial);
    vq_loss_kernel<<<1, 256, 0, stream>>>(partial, NN / 256, out + (size_t)NN * CC);
}
